// DeepSeekMLHA_18425409700060
// MI455X (gfx1250) — matrix operand order repaired and checked by reading, not compiled, not run on hardware
//
#include <hip/hip_runtime.h>
#include <hip/hip_bf16.h>

// ---------------------------------------------------------------------------
// DeepSeek MLA forward for MI455X (gfx1250).
// wave32 + v_wmma_f32_16x16x32_bf16, async global->LDS tile staging (ASYNCcnt)
// with double buffering; fragment reads are contiguous 32B per lane
// (lowered to ds_load_b128 / global_load_b128); LDS fragment streams are
// software-pipelined depth-2 so DS latency overlaps WMMA issue.
// ---------------------------------------------------------------------------

typedef __attribute__((ext_vector_type(16))) __bf16 v16bf;
typedef __attribute__((ext_vector_type(8)))  float  v8f;
typedef __attribute__((ext_vector_type(4)))  int    v4i;

union Frag16 {
    v16bf        v;
    unsigned int u[8];
    __bf16       h[16];
};

// ---- async global->LDS (CDNA5 GLOBAL_LOAD_ASYNC_TO_LDS_B128), guarded ------
#if defined(__has_builtin)
#if __has_builtin(__builtin_amdgcn_global_load_async_to_lds_b128) && \
    __has_builtin(__builtin_amdgcn_s_wait_asynccnt)
#define USE_ASYNC_LDS 1
#endif
#endif
#ifndef USE_ASYNC_LDS
#define USE_ASYNC_LDS 0
#endif

typedef __attribute__((address_space(1))) v4i as1_v4i;   // global (__device__)
typedef __attribute__((address_space(3))) v4i as3_v4i;   // LDS (__shared__)

__device__ __forceinline__ void copy16_g2l(const __bf16* g, __bf16* l) {
#if USE_ASYNC_LDS
    __builtin_amdgcn_global_load_async_to_lds_b128((as1_v4i*)g, (as3_v4i*)l,
                                                   0, 0);
#else
    *(uint4*)l = *(const uint4*)g;
#endif
}

template <int N>
__device__ __forceinline__ void wait_async() {
#if USE_ASYNC_LDS
    __builtin_amdgcn_s_wait_asynccnt((short)N);
#endif
}

// ---------------------------------------------------------------------------
// fp32 -> bf16 elementwise (grid-stride) — for x (stays row-major, A operand)
// ---------------------------------------------------------------------------
__global__ void f32_to_bf16_kernel(const float* __restrict__ in,
                                   __bf16* __restrict__ out, size_t n) {
    size_t i = (size_t)blockIdx.x * blockDim.x + threadIdx.x;
    size_t stride = (size_t)gridDim.x * blockDim.x;
    for (; i < n; i += stride) out[i] = (__bf16)in[i];
}

// ---------------------------------------------------------------------------
// fused transpose + fp32->bf16: in[R][C] f32 -> out[C][R] bf16 (32x32 tiles)
// block (32,8); R,C multiples of 32
// ---------------------------------------------------------------------------
__global__ __launch_bounds__(256)
void transpose_f32_bf16_kernel(const float* __restrict__ in,
                               __bf16* __restrict__ out, int R, int C) {
    __shared__ float tile[32][33];
    const int c0 = blockIdx.x * 32, r0 = blockIdx.y * 32;
    const int tx = threadIdx.x, ty = threadIdx.y;
    #pragma unroll
    for (int i = 0; i < 32; i += 8)
        tile[ty + i][tx] = in[(size_t)(r0 + ty + i) * C + c0 + tx];
    __syncthreads();
    #pragma unroll
    for (int i = 0; i < 32; i += 8)
        out[(size_t)(c0 + ty + i) * R + r0 + tx] = (__bf16)tile[tx][ty + i];
}

// ---------------------------------------------------------------------------
// bf16 strided transpose: out[c*ostride + r] = in[r*istride + c]
// ---------------------------------------------------------------------------
__global__ __launch_bounds__(256)
void transpose_bf16_kernel(const __bf16* __restrict__ in, size_t istride,
                           __bf16* __restrict__ out, size_t ostride,
                           int R, int C) {
    __shared__ __bf16 tile[32][34];
    const int c0 = blockIdx.x * 32, r0 = blockIdx.y * 32;
    const int tx = threadIdx.x, ty = threadIdx.y;
    #pragma unroll
    for (int i = 0; i < 32; i += 8)
        tile[ty + i][tx] = in[(size_t)(r0 + ty + i) * istride + c0 + tx];
    __syncthreads();
    #pragma unroll
    for (int i = 0; i < 32; i += 8)
        out[(size_t)(c0 + ty + i) * ostride + r0 + tx] = tile[tx][ty + i];
}

// ---------------------------------------------------------------------------
// RMSNorm over rank=256 rows (one 256-thread block per row)
// ---------------------------------------------------------------------------
__global__ __launch_bounds__(256)
void rmsnorm_kernel(const float* __restrict__ in, const float* __restrict__ w,
                    __bf16* __restrict__ out, int rank) {
    __shared__ float red[8];
    const int row = blockIdx.x;
    const int tid = threadIdx.x;
    float v = in[(size_t)row * rank + tid];
    float s = v * v;
    #pragma unroll
    for (int m = 1; m < 32; m <<= 1) s += __shfl_xor(s, m, 32);
    if ((tid & 31) == 0) red[tid >> 5] = s;
    __syncthreads();
    float tot = 0.f;
    #pragma unroll
    for (int i = 0; i < 8; ++i) tot += red[i];
    float r = rsqrtf(tot / (float)rank + 1e-6f);
    out[(size_t)row * rank + tid] = (__bf16)(w[tid] * v * r);
}

// ---------------------------------------------------------------------------
// Tiled bf16 GEMM with B pre-transposed: C[M,N] = A[M,K] * BT[N,K]^T.
// Block 128x128, K-step 32, 8 waves (4Mx2N), each wave 2x4 WMMA tiles.
// Double-buffered LDS, async global->LDS staging.
// Requires M,N % 128 == 0, K % 32 == 0 (true for all calls here).
// ---------------------------------------------------------------------------
#define GBM 128
#define GBN 128
#define GBK 32

template <bool OUT_BF16>
__global__ __launch_bounds__(256)
void gemm_bf16_wmma(const __bf16* __restrict__ A, const __bf16* __restrict__ BT,
                    void* __restrict__ C, int M, int N, int K) {
    __shared__ __align__(16) __bf16 As[2][GBM * GBK];   // 2 x 8 KB
    __shared__ __align__(16) __bf16 Bs[2][GBN * GBK];   // 2 x 8 KB

    const int tid  = threadIdx.x;
    const int wid  = tid >> 5;
    const int lane = tid & 31;
    const int wm   = wid & 3;     // 0..3 : 32-row strip
    const int wn   = wid >> 2;    // 0..1 : 64-col strip
    const int m0   = blockIdx.y * GBM;
    const int n0   = blockIdx.x * GBN;
    const int fm   = lane & 15;
    const int kh   = lane >> 4;

    // cooperative staging: thread t moves 32B of A and 32B of BT per K-step
    const int crow  = tid >> 1;          // 0..127
    const int chalf = (tid & 1) << 4;    // 0 or 16 bf16

    v8f acc[2][4];
    const v8f vzero = {0.f, 0.f, 0.f, 0.f, 0.f, 0.f, 0.f, 0.f};
    #pragma unroll
    for (int mt = 0; mt < 2; ++mt)
        #pragma unroll
        for (int nt = 0; nt < 4; ++nt) acc[mt][nt] = vzero;

    auto issue = [&](int k0, int buf) {
        const __bf16* ga = A + (size_t)(m0 + crow) * K + k0 + chalf;
        __bf16* la = &As[buf][crow * GBK + chalf];
        copy16_g2l(ga, la);
        copy16_g2l(ga + 8, la + 8);
        const __bf16* gb = BT + (size_t)(n0 + crow) * K + k0 + chalf;
        __bf16* lb = &Bs[buf][crow * GBK + chalf];
        copy16_g2l(gb, lb);
        copy16_g2l(gb + 8, lb + 8);
    };

    issue(0, 0);
    const int nsteps = K / GBK;
    for (int s = 0; s < nsteps; ++s) {
        const int buf = s & 1;
        const bool more = (s + 1 < nsteps);
        if (more) { issue((s + 1) * GBK, buf ^ 1); wait_async<4>(); }
        else      { wait_async<0>(); }
        __syncthreads();

        // A fragments: contiguous 32B of K per lane (k = 16*kh + s)
        Frag16 af[2];
        #pragma unroll
        for (int mt = 0; mt < 2; ++mt) {
            const __bf16* src = &As[buf][(wm * 32 + mt * 16 + fm) * GBK];
            #pragma unroll
            for (int j = 0; j < 8; ++j) {
                const int kk = kh * 16 + 2 * j;
                af[mt].u[j] = *(const unsigned int*)(src + kk);
            }
        }
        // B fragments from BT: lane = output column, contiguous 32B of K
        Frag16 bfm[4];
        #pragma unroll
        for (int nt = 0; nt < 4; ++nt) {
            const __bf16* src =
                &Bs[buf][(wn * 64 + nt * 16 + fm) * GBK + kh * 16];
            #pragma unroll
            for (int j = 0; j < 8; ++j)
                bfm[nt].u[j] = *(const unsigned int*)(src + 2 * j);
        }
        #pragma unroll
        for (int mt = 0; mt < 2; ++mt)
            #pragma unroll
            for (int nt = 0; nt < 4; ++nt)
                acc[mt][nt] = __builtin_amdgcn_wmma_f32_16x16x32_bf16(
                    false, af[mt].v, false, bfm[nt].v, (short)0, acc[mt][nt],
                    false, false);
        __syncthreads();
    }

    // epilogue: C layout row = r + 8*kh, col = fm
    #pragma unroll
    for (int mt = 0; mt < 2; ++mt)
        #pragma unroll
        for (int nt = 0; nt < 4; ++nt) {
            const int col = n0 + wn * 64 + nt * 16 + fm;
            #pragma unroll
            for (int r = 0; r < 8; ++r) {
                const int row = m0 + wm * 32 + mt * 16 + r + kh * 8;
                const float v = acc[mt][nt][r];
                if (OUT_BF16)
                    ((__bf16*)C)[(size_t)row * N + col] = (__bf16)v;
                else
                    ((float*)C)[(size_t)row * N + col] = v;
            }
        }
}

// ---------------------------------------------------------------------------
// Flash-style causal MLA attention.
//   q : bf16 [B*L, 2048]   (head h at col h*128)
//   kv: bf16 [B*L, 4096]   (k half, cols [0,2048))
//   vt: bf16 [B][2048][L]  (v transposed: row = h*128+d, col = token)
//   o : bf16 [B*L, 2048]
// Grid (L/64, B*16), block 128 = 4 waves; all waves share head + key chunks,
// so K/V chunks are staged once per block into double-buffered LDS (async).
// Fragment streams from LDS are depth-2 pipelined (issue t+1, WMMA t).
// ---------------------------------------------------------------------------
__global__ __launch_bounds__(128)
void mla_attention_kernel(const __bf16* __restrict__ q,
                          const __bf16* __restrict__ kv,
                          const __bf16* __restrict__ vt,
                          __bf16* __restrict__ o, int L) {
    __shared__ __align__(16) __bf16 Ks[2][32 * 128];  // [key][d]   2 x 8 KB
    __shared__ __align__(16) __bf16 Vs[2][128 * 32];  // [d][key]   2 x 8 KB
    __shared__ __align__(16) __bf16 pscr[4 * 16 * 32];

    const int tid  = threadIdx.x;
    const int wid  = tid >> 5;
    const int lane = tid & 31;
    const int bh   = blockIdx.y;
    const int b    = bh >> 4;
    const int h    = bh & 15;
    const int lq   = blockIdx.x * 64 + wid * 16;
    const int fm   = lane & 15;
    const int kh   = lane >> 4;
    const float scale = 0.08838834764831845f;   // 1/sqrt(128)

    // Q fragments: 4 d-chunks of 32 (contiguous per lane)
    Frag16 qf[4];
    {
        const __bf16* qrow = q + ((size_t)(b * L + lq + fm)) * 2048 + h * 128;
        #pragma unroll
        for (int dc = 0; dc < 4; ++dc)
            #pragma unroll
            for (int j = 0; j < 8; ++j) {
                const int dd = dc * 32 + kh * 16 + 2 * j;
                qf[dc].u[j] = *(const unsigned int*)(qrow + dd);
            }
    }

    float mrow[8], lrow[8];
    v8f O[8];
    const v8f vzero = {0.f, 0.f, 0.f, 0.f, 0.f, 0.f, 0.f, 0.f};
    #pragma unroll
    for (int r = 0; r < 8; ++r) { mrow[r] = -3.0e38f; lrow[r] = 0.f; }
    #pragma unroll
    for (int dc = 0; dc < 8; ++dc) O[dc] = vzero;

    // cooperative K/V chunk staging (128 threads, 8 async b128 each)
    auto issue = [&](int kb, int buf) {
        #pragma unroll
        for (int i = 0; i < 4; ++i) {               // K: 32 rows x 256 B
            const int slot = tid + i * 128;
            const int row = slot >> 4, seg = (slot & 15) * 8;
            copy16_g2l(kv + ((size_t)(b * L + kb + row)) * 4096 + h * 128 + seg,
                       &Ks[buf][row * 128 + seg]);
        }
        #pragma unroll
        for (int i = 0; i < 4; ++i) {               // V^T: 128 rows x 64 B
            const int slot = tid + i * 128;
            const int row = slot >> 2, seg = (slot & 3) * 8;
            copy16_g2l(vt + ((size_t)(b * 2048 + h * 128 + row)) * (size_t)L +
                           kb + seg,
                       &Vs[buf][row * 32 + seg]);
        }
    };

    const int nchunks = (blockIdx.x * 64 + 64) >> 5;   // block-uniform
    __bf16* myscr = pscr + wid * 16 * 32;

    issue(0, 0);
    for (int ci = 0; ci < nchunks; ++ci) {
        const int kb = ci * 32;
        const int buf = ci & 1;
        const bool more = (ci + 1 < nchunks);
        if (more) { issue(kb + 32, buf ^ 1); wait_async<8>(); }
        else      { wait_async<0>(); }
        __syncthreads();

        // ---- scores S = Q.K^T: flattened (kk,dc) stream, depth-2 pipeline --
        v8f S[2];
        S[0] = vzero; S[1] = vzero;
        {
            auto load_kf = [&](Frag16& f, int t) {   // t = kk*4 + dc
                const int kk = t >> 2, dc = t & 3;
                const __bf16* src =
                    &Ks[buf][(kk * 16 + fm) * 128 + dc * 32 + kh * 16];
                #pragma unroll
                for (int j = 0; j < 8; ++j)
                    f.u[j] = *(const unsigned int*)(src + 2 * j);
            };
            Frag16 kf2[2];
            load_kf(kf2[0], 0);
            #pragma unroll
            for (int t = 0; t < 8; ++t) {
                if (t < 7) load_kf(kf2[(t + 1) & 1], t + 1);
                const int kk = t >> 2, dc = t & 3;
                S[kk] = __builtin_amdgcn_wmma_f32_16x16x32_bf16(
                    false, qf[dc].v, false, kf2[t & 1].v, (short)0, S[kk],
                    false, false);
            }
        }
        // ---- scale + causal mask ----
        #pragma unroll
        for (int kk = 0; kk < 2; ++kk)
            #pragma unroll
            for (int r = 0; r < 8; ++r) {
                const int qg = lq + r + kh * 8;
                const int kg = kb + kk * 16 + fm;
                const float v = S[kk][r] * scale;
                S[kk][r] = (kg <= qg) ? v : -3.0e38f;
            }
        // ---- online softmax (row reductions over 16-lane groups) ----
        #pragma unroll
        for (int r = 0; r < 8; ++r) {
            float t = fmaxf(S[0][r], S[1][r]);
            t = fmaxf(t, __shfl_xor(t, 1, 32));
            t = fmaxf(t, __shfl_xor(t, 2, 32));
            t = fmaxf(t, __shfl_xor(t, 4, 32));
            t = fmaxf(t, __shfl_xor(t, 8, 32));
            const float mnew = fmaxf(mrow[r], t);
            const float esc  = __expf(mrow[r] - mnew);
            const float p0 = __expf(S[0][r] - mnew);
            const float p1 = __expf(S[1][r] - mnew);
            S[0][r] = p0; S[1][r] = p1;
            float rs = p0 + p1;
            rs += __shfl_xor(rs, 1, 32);
            rs += __shfl_xor(rs, 2, 32);
            rs += __shfl_xor(rs, 4, 32);
            rs += __shfl_xor(rs, 8, 32);
            lrow[r] = lrow[r] * esc + rs;
            mrow[r] = mnew;
            #pragma unroll
            for (int dc = 0; dc < 8; ++dc) O[dc][r] *= esc;
        }
        // ---- transpose P (C layout) -> A-fragment layout via LDS ----
        #pragma unroll
        for (int kk = 0; kk < 2; ++kk)
            #pragma unroll
            for (int r = 0; r < 8; ++r)
                myscr[(r + kh * 8) * 32 + kk * 16 + fm] = (__bf16)S[kk][r];
        __syncthreads();
        Frag16 pf;
        #pragma unroll
        for (int j = 0; j < 8; ++j) {
            const int kkk = kh * 16 + 2 * j;
            pf.u[j] = *(const unsigned int*)(myscr + fm * 32 + kkk);
        }
        __syncthreads();
        // ---- O += P.V: depth-2 pipelined V-fragment stream ----
        {
            auto load_vf = [&](Frag16& f, int dc) {
                const __bf16* src = &Vs[buf][(dc * 16 + fm) * 32 + kh * 16];
                #pragma unroll
                for (int j = 0; j < 8; ++j)
                    f.u[j] = *(const unsigned int*)(src + 2 * j);
            };
            Frag16 vf2[2];
            load_vf(vf2[0], 0);
            #pragma unroll
            for (int dc = 0; dc < 8; ++dc) {
                if (dc < 7) load_vf(vf2[(dc + 1) & 1], dc + 1);
                O[dc] = __builtin_amdgcn_wmma_f32_16x16x32_bf16(
                    false, pf.v, false, vf2[dc & 1].v, (short)0, O[dc],
                    false, false);
            }
        }
        __syncthreads();   // protect buf before it is re-staged
    }

    // ---- normalize + store ----
    __bf16* orow = o + ((size_t)(b * L + lq)) * 2048 + h * 128;
    #pragma unroll
    for (int r = 0; r < 8; ++r) {
        const float inv = 1.0f / lrow[r];
        const int row = r + kh * 8;
        #pragma unroll
        for (int dc = 0; dc < 8; ++dc)
            orow[(size_t)row * 2048 + dc * 16 + fm] = (__bf16)(O[dc][r] * inv);
    }
}

// ---------------------------------------------------------------------------
// host launcher
// ---------------------------------------------------------------------------
extern "C" void kernel_launch(void* const* d_in, const int* in_sizes, int n_in,
                              void* d_out, int out_size, void* d_ws, size_t ws_size,
                              hipStream_t stream) {
    (void)in_sizes; (void)n_in; (void)out_size; (void)ws_size;

    const float* x   = (const float*)d_in[0];   // [2,2048,2048]
    const float* Wq  = (const float*)d_in[1];   // [2048,2048]
    const float* Wkd = (const float*)d_in[2];   // [2048,256]
    const float* wn  = (const float*)d_in[3];   // [256]
    const float* Wku = (const float*)d_in[4];   // [256,4096]
    const float* Wo  = (const float*)d_in[5];   // [2048,2048]

    const int Lt = 2048, Bt = 2;
    const int M  = Bt * Lt;      // 4096 tokens
    const int D  = 2048;         // hidden / attn dim
    const int R  = 256;          // kv rank
    const int A2 = 4096;         // 2*attn dim

    char* p = (char*)d_ws;
    __bf16* x_b   = (__bf16*)p; p += (size_t)M * D * 2;     // 16 MB
    __bf16* WqT   = (__bf16*)p; p += (size_t)D * D * 2;     // 8 MB  [N][K]
    __bf16* WkdT  = (__bf16*)p; p += (size_t)D * R * 2;     // 1 MB  [R][D]
    __bf16* WkuT  = (__bf16*)p; p += (size_t)R * A2 * 2;    // 2 MB  [A2][R]
    __bf16* WoT   = (__bf16*)p; p += (size_t)D * D * 2;     // 8 MB
    __bf16* q_b   = (__bf16*)p; p += (size_t)M * D * 2;     // 16 MB
    float*  lat_f = (float*)p;  p += (size_t)M * R * 4;     // 4 MB
    __bf16* lat_b = (__bf16*)p; p += (size_t)M * R * 2;     // 2 MB
    __bf16* kv_b  = (__bf16*)p; p += (size_t)M * A2 * 2;    // 32 MB
    __bf16* vt_b  = (__bf16*)p; p += (size_t)M * D * 2;     // 16 MB [B][D][L]
    __bf16* at_b  = (__bf16*)p; p += (size_t)M * D * 2;     // 16 MB

    // x -> bf16 row-major; weights -> bf16 transposed [out][in]
    f32_to_bf16_kernel<<<1024, 256, 0, stream>>>(x, x_b, (size_t)M * D);
    const dim3 tb(32, 8);
    transpose_f32_bf16_kernel<<<dim3(D / 32, D / 32), tb, 0, stream>>>(Wq, WqT, D, D);
    transpose_f32_bf16_kernel<<<dim3(R / 32, D / 32), tb, 0, stream>>>(Wkd, WkdT, D, R);
    transpose_f32_bf16_kernel<<<dim3(A2 / 32, R / 32), tb, 0, stream>>>(Wku, WkuT, R, A2);
    transpose_f32_bf16_kernel<<<dim3(D / 32, D / 32), tb, 0, stream>>>(Wo, WoT, D, D);

    const dim3 blk(256);
    // q = x @ Wq  -> bf16
    gemm_bf16_wmma<true><<<dim3(D / 128, M / 128), blk, 0, stream>>>(
        x_b, WqT, q_b, M, D, D);
    // latent_raw = x @ Wkv_down -> f32
    gemm_bf16_wmma<false><<<dim3(R / 128, M / 128), blk, 0, stream>>>(
        x_b, WkdT, lat_f, M, R, D);
    // rmsnorm -> bf16
    rmsnorm_kernel<<<M, 256, 0, stream>>>(lat_f, wn, lat_b, R);
    // kv = latent @ Wkv_up -> bf16 (k | v per row)
    gemm_bf16_wmma<true><<<dim3(A2 / 128, M / 128), blk, 0, stream>>>(
        lat_b, WkuT, kv_b, M, A2, R);
    // v half -> per-batch transpose [D][L] for contiguous V fragments
    for (int b = 0; b < Bt; ++b)
        transpose_bf16_kernel<<<dim3(D / 32, Lt / 32), tb, 0, stream>>>(
            kv_b + (size_t)b * Lt * A2 + D, (size_t)A2,
            vt_b + (size_t)b * D * Lt, (size_t)Lt, Lt, D);
    // causal attention
    mla_attention_kernel<<<dim3(Lt / 64, Bt * 16), 128, 0, stream>>>(
        q_b, kv_b, vt_b, at_b, Lt);
    // out = attn @ Wo -> f32
    gemm_bf16_wmma<false><<<dim3(D / 128, M / 128), blk, 0, stream>>>(
        at_b, WoT, d_out, M, D, D);
}
